// ColorNetwork_32495722561720
// MI455X (gfx1250) — matrix operand order repaired and checked by reading, not compiled, not run on hardware
//
#include <hip/hip_runtime.h>
#include <hip/hip_bf16.h>

// MI455X / gfx1250: wave32, WMMA bf16 16x16x32 with f32 accumulate,
// TDM (tensor_load_to_lds) for weight staging.

typedef __attribute__((ext_vector_type(16))) __bf16 v16bf;
typedef __attribute__((ext_vector_type(8)))  float  v8f;
typedef __attribute__((ext_vector_type(4)))  unsigned int v4u_;
typedef __attribute__((ext_vector_type(8)))  int v8i_;
typedef __attribute__((ext_vector_type(4)))  int v4i_;

#define FLOAT_MIN_F (-3.4028234663852886e+38f)

#define BM 128                      // batch rows per block (8 waves x 16 rows)
#define KDIM 256                    // all layers have K = 256
#define WSTRIDE (KDIM + 16)         // LDS weight column stride (bf16), keeps b128 alignment
#define ASTRIDE (256 + 8)           // LDS activation row stride (bf16), 16B-aligned rows
#define LDS_W_ELEMS (256 * WSTRIDE) // max N = 256 columns
#define LDS_A_ELEMS (BM * ASTRIDE)
#define LDS_BYTES ((size_t)(LDS_W_ELEMS + LDS_A_ELEMS) * 2)

// pre-converted weight buffer layout in d_ws (bf16 elements)
#define PRE_W1_OFF 0
#define PRE_W2_OFF (256 * WSTRIDE)
#define PRE_W3_OFF (2 * 256 * WSTRIDE)
#define PRE_TOTAL_ELEMS (2 * 256 * WSTRIDE + 128 * WSTRIDE)
#define PRE_TOTAL_BYTES ((size_t)PRE_TOTAL_ELEMS * 2)

#if defined(__HIP_DEVICE_COMPILE__) && __has_builtin(__builtin_amdgcn_tensor_load_to_lds)
#define HAVE_TDM 1
#else
#define HAVE_TDM 0
#endif

union Frag {
    v16bf v;
    uint4 q[2];
};

__device__ __forceinline__ unsigned short f2bf_bits(float f) {
    union { float f; unsigned u; } in;
    in.f = f;
    unsigned u = in.u + 0x7fffu + ((in.u >> 16) & 1u);   // round-to-nearest-even
    return (unsigned short)(u >> 16);
}

__device__ __forceinline__ __bf16 f2bf(float f) {
    union { unsigned short s; __bf16 h; } out;
    out.s = f2bf_bits(f);
    return out.h;
}

__device__ __forceinline__ v8f wmma_bf16(v16bf a, v16bf b, v8f c) {
    // (neg_a, A, neg_b, B, c_mod, C, reuse_a, reuse_b)
    return __builtin_amdgcn_wmma_f32_16x16x32_bf16(false, a, false, b, (short)0, c,
                                                   false, false);
}

#if HAVE_TDM
// One-shot TDM DMA: copy `nbytes` (multiple of 8, < 512KB) from global `src`
// into LDS at byte offset `lds_off`. 1-D tile, data_size = 8 bytes.
__device__ __forceinline__ void tdm_copy_to_lds(const void* src, unsigned lds_off,
                                                unsigned nbytes) {
    const unsigned units = nbytes >> 3;                // 8-byte elements (< 65536)
    const unsigned long long ga = (unsigned long long)src;
    v4u_ g0;
    g0.x = 1u;                                         // count=1, user mode, no gather
    g0.y = lds_off;                                    // D#.lds_addr (bytes)
    g0.z = (unsigned)ga;                               // global_addr[31:0]
    g0.w = (unsigned)((ga >> 32) & 0x01ffffffull)      // global_addr[56:32]
           | (2u << 30);                               // type = 2 ("image")
    v8i_ g1;
    g1[0] = (int)(3u << 16);                           // data_size=8B, mask=0, no iter/pad
    g1[1] = (int)(units << 16);                        // tensor_dim0[15:0] (bits 63:48)
    g1[2] = (int)(1u << 16);                           // tensor_dim0 hi=0, tensor_dim1=1
    g1[3] = (int)(units << 16);                        // tile_dim0 (bits 127:112)
    g1[4] = 1;                                         // tile_dim1=1, tile_dim2=0
    g1[5] = (int)units;                                // tensor_dim0_stride[31:0]
    g1[6] = 0;                                         // stride hi / dim1_stride lo
    g1[7] = 0;
    v4i_ z4 = {0, 0, 0, 0};                            // groups 2/3: dims unused
    v8i_ z8 = {0, 0, 0, 0, 0, 0, 0, 0};
    // 6-arg form (clang-23 / therock headers): (g0, g1, g2, g3, extra, cpol)
    __builtin_amdgcn_tensor_load_to_lds(g0, g1, z4, z4, z8, 0);
}
#endif

// Fallback: stage a K x N fp32 weight matrix (row-major, K=256) into LDS as bf16,
// column-major with padded stride (used when no pre-converted scratch is available).
__device__ __forceinline__ void stage_weights_cvt(const float* __restrict__ W, int N,
                                                  __bf16* __restrict__ wlds, int tid) {
    const int total = KDIM * N;
    for (int i = tid; i < total; i += 256) {
        int k = i / N;
        int n = i - k * N;                 // global read coalesced over n
        wlds[n * WSTRIDE + k] = f2bf(W[i]);
    }
}

// Fallback copy of pre-converted bf16 weights when TDM builtin is unavailable.
__device__ __forceinline__ void stage_weights_copy(const unsigned short* __restrict__ src,
                                                   int N, __bf16* __restrict__ wlds,
                                                   int tid) {
    const int words = (N * WSTRIDE) / 8;   // uint4 chunks (16B)
    for (int i = tid; i < words; i += 256)
        ((uint4*)wlds)[i] = ((const uint4*)src)[i];
}

// ---------------------------------------------------------------------------
// Prep kernel: fp32 row-major [256 x N] -> bf16 column-major padded (WSTRIDE)
// ---------------------------------------------------------------------------
__global__ __launch_bounds__(256)
void convert_weights_kernel(const float* __restrict__ W,
                            unsigned short* __restrict__ dst, int N) {
    int i = blockIdx.x * 256 + threadIdx.x;        // i enumerates (n,k), k fastest
    if (i < KDIM * N) {
        int k = i & (KDIM - 1);
        int n = i >> 8;
        dst[n * WSTRIDE + k] = f2bf_bits(W[k * N + n]);  // contiguous dst stores
    }
}

// ---------------------------------------------------------------------------
// Main fused MLP kernel
// ---------------------------------------------------------------------------
__global__ __launch_bounds__(256)
void mlp3_mask_kernel(const float* __restrict__ obs,
                      const int*   __restrict__ mask,
                      const float* __restrict__ W1, const float* __restrict__ b1,
                      const float* __restrict__ W2, const float* __restrict__ b2,
                      const float* __restrict__ W3, const float* __restrict__ b3,
                      const unsigned short* __restrict__ wpre,   // may be null
                      float* __restrict__ out) {
    extern __shared__ char smem_raw[];
    __bf16* wlds = (__bf16*)smem_raw;              // 256 x WSTRIDE bf16 (139.3 KB)
    __bf16* act  = wlds + LDS_W_ELEMS;             // BM  x ASTRIDE bf16 ( 67.6 KB)

    const int tid   = threadIdx.x;
    const int wave  = tid >> 5;
    const int lane  = tid & 31;
    const int l15   = lane & 15;
    const int khalf = lane >> 4;

    const unsigned lds_w_off = (unsigned)__builtin_amdgcn_groupstaticsize();

    // ---- Stage W1 (TDM DMA overlapped with X conversion below) ----
    if (wpre) {
#if HAVE_TDM
        if (wave == 0)
            tdm_copy_to_lds(wpre + PRE_W1_OFF, lds_w_off, 256 * WSTRIDE * 2);
#else
        stage_weights_copy(wpre + PRE_W1_OFF, 256, wlds, tid);
#endif
    } else {
        stage_weights_cvt(W1, 256, wlds, tid);
    }

    // ---- Stage input tile X[BM x 256] (fp32 -> bf16, row-major in LDS) ----
    {
        const float* xblk = obs + (size_t)blockIdx.x * BM * 256;
        for (int i = tid; i < BM * 128; i += 256) {   // float2 pairs
            int r  = i >> 7;
            int cp = i & 127;
            float2 v = *(const float2*)(xblk + r * 256 + cp * 2);
            union { unsigned short s[2]; unsigned u; } pk;
            pk.s[0] = f2bf_bits(v.x);
            pk.s[1] = f2bf_bits(v.y);
            *(unsigned*)(act + r * ASTRIDE + cp * 2) = pk.u;
        }
    }
#if HAVE_TDM
    if (wpre && wave == 0) __builtin_amdgcn_s_wait_tensorcnt(0);
#endif
    __syncthreads();

    const int mrow  = wave * 16 + l15;        // A-fragment row for this lane
    const int mbase = wave * 16 + khalf * 8;  // first C/D row for this lane

    // ---- Layers 1 and 2: act = relu(act @ W + b), N = 256 (16 tiles) ----
    #pragma unroll 1
    for (int layer = 0; layer < 2; ++layer) {
        const float* bias = (layer == 0) ? b1 : b2;

        Frag a[8];
        const __bf16* arow = act + mrow * ASTRIDE + khalf * 8;
        #pragma unroll
        for (int t = 0; t < 8; ++t) {
            a[t].q[0] = *(const uint4*)(arow + t * 32);
            a[t].q[1] = *(const uint4*)(arow + t * 32 + 16);
        }

        #pragma unroll 1
        for (int n = 0; n < 16; ++n) {
            v8f c = {0.f, 0.f, 0.f, 0.f, 0.f, 0.f, 0.f, 0.f};
            const __bf16* bcol = wlds + (n * 16 + l15) * WSTRIDE + khalf * 8;
            #pragma unroll
            for (int t = 0; t < 8; ++t) {
                Frag b;
                b.q[0] = *(const uint4*)(bcol + t * 32);
                b.q[1] = *(const uint4*)(bcol + t * 32 + 16);
                c = wmma_bf16(a[t].v, b.v, c);
            }
            const float bv = bias[n * 16 + l15];
            #pragma unroll
            for (int j = 0; j < 8; ++j) {
                float v = fmaxf(c[j] + bv, 0.f);
                act[(mbase + j) * ASTRIDE + n * 16 + l15] = f2bf(v);
            }
        }

        __syncthreads();                         // all waves done with wlds
        if (wpre) {
#if HAVE_TDM
            if (wave == 0) {
                if (layer == 0)
                    tdm_copy_to_lds(wpre + PRE_W2_OFF, lds_w_off, 256 * WSTRIDE * 2);
                else
                    tdm_copy_to_lds(wpre + PRE_W3_OFF, lds_w_off, 128 * WSTRIDE * 2);
                __builtin_amdgcn_s_wait_tensorcnt(0);
            }
#else
            if (layer == 0) stage_weights_copy(wpre + PRE_W2_OFF, 256, wlds, tid);
            else            stage_weights_copy(wpre + PRE_W3_OFF, 128, wlds, tid);
#endif
        } else {
            if (layer == 0) stage_weights_cvt(W2, 256, wlds, tid);
            else            stage_weights_cvt(W3, 128, wlds, tid);
        }
        __syncthreads();
    }

    // ---- Layer 3: logits = act @ W3 + b3, N = 128 (8 tiles) + masking ----
    {
        Frag a[8];
        const __bf16* arow = act + mrow * ASTRIDE + khalf * 8;
        #pragma unroll
        for (int t = 0; t < 8; ++t) {
            a[t].q[0] = *(const uint4*)(arow + t * 32);
            a[t].q[1] = *(const uint4*)(arow + t * 32 + 16);
        }

        const size_t rowg0 = (size_t)blockIdx.x * BM + (size_t)mbase;
        unsigned vbits = 0;                  // bit j: row (mbase+j) has a valid col here

        #pragma unroll 1
        for (int n = 0; n < 8; ++n) {
            v8f c = {0.f, 0.f, 0.f, 0.f, 0.f, 0.f, 0.f, 0.f};
            const __bf16* bcol = wlds + (n * 16 + l15) * WSTRIDE + khalf * 8;
            #pragma unroll
            for (int t = 0; t < 8; ++t) {
                Frag b;
                b.q[0] = *(const uint4*)(bcol + t * 32);
                b.q[1] = *(const uint4*)(bcol + t * 32 + 16);
                c = wmma_bf16(a[t].v, b.v, c);
            }
            const float bv  = b3[n * 16 + l15];
            const int   col = n * 16 + l15;
            #pragma unroll
            for (int j = 0; j < 8; ++j) {
                size_t idx = (rowg0 + j) * 128 + col;
                int mv = mask[idx];
                float v = c[j] + bv;
                if (mv) vbits |= (1u << j); else v = FLOAT_MIN_F;
                out[idx] = v;
            }
        }

        // OR validity across the 16 lanes sharing each row (xor<16 stays in half)
        #pragma unroll
        for (int off = 1; off < 16; off <<= 1)
            vbits |= (unsigned)__shfl_xor((int)vbits, off, 32);

        if (l15 == 0) {                      // lanes 0 and 16 cover all 16 rows
            #pragma unroll
            for (int j = 0; j < 8; ++j)
                if (!((vbits >> j) & 1u))
                    out[(rowg0 + j) * 128] = 1.0f;   // same-lane store order is kept
        }
    }
}

extern "C" void kernel_launch(void* const* d_in, const int* in_sizes, int n_in,
                              void* d_out, int out_size, void* d_ws, size_t ws_size,
                              hipStream_t stream) {
    const float* obs  = (const float*)d_in[0];
    const int*   mask = (const int*)d_in[1];
    const float* W1   = (const float*)d_in[2];
    const float* b1   = (const float*)d_in[3];
    const float* W2   = (const float*)d_in[4];
    const float* b2   = (const float*)d_in[5];
    const float* W3   = (const float*)d_in[6];
    const float* b3   = (const float*)d_in[7];
    float* out = (float*)d_out;

    const int B = in_sizes[0] / 256;             // 65536
    (void)n_in; (void)out_size;

    const bool use_ws = (d_ws != nullptr) && (ws_size >= PRE_TOTAL_BYTES);
    unsigned short* wpre = use_ws ? (unsigned short*)d_ws : nullptr;

    if (use_ws) {
        convert_weights_kernel<<<256, 256, 0, stream>>>(W1, wpre + PRE_W1_OFF, 256);
        convert_weights_kernel<<<256, 256, 0, stream>>>(W2, wpre + PRE_W2_OFF, 256);
        convert_weights_kernel<<<128, 256, 0, stream>>>(W3, wpre + PRE_W3_OFF, 128);
    }

    (void)hipFuncSetAttribute(reinterpret_cast<const void*>(&mlp3_mask_kernel),
                              hipFuncAttributeMaxDynamicSharedMemorySize,
                              (int)LDS_BYTES);

    mlp3_mask_kernel<<<B / BM, 256, LDS_BYTES, stream>>>(obs, mask, W1, b1, W2, b2,
                                                         W3, b3, wpre, out);
}
